// MambaLayer_4337916969264
// MI455X (gfx1250) — compile-verified, matrix operand order checked by reading, not run on hardware
//
#include <hip/hip_runtime.h>
#include <hip/hip_bf16.h>

// ---------------------------------------------------------------------------
// Mamba layer for MI455X (gfx1250, wave32, WMMA + TDM, double-buffered LDS).
// B=2, L=2048, D_MODEL=2048, D_INNER=4096, N=16, DT_RANK=128, K=4
// ---------------------------------------------------------------------------

#define B_ 2
#define L_ 2048
#define DMODEL 2048
#define DINNER 4096
#define NSTATE 16
#define DTRANK 128
#define KCONV 4
#define ROWS (B_ * L_)                  // 4096 token rows
#define XPROJ_N (DTRANK + 2 * NSTATE)   // 160 (logical)
#define XPROJ_PAD 192                   // padded to a BN multiple

typedef __attribute__((ext_vector_type(16))) __bf16 v16bf;
typedef __attribute__((ext_vector_type(8)))  __bf16 v8bf;
typedef __attribute__((ext_vector_type(8)))  float  v8f;
typedef __attribute__((ext_vector_type(4)))  unsigned u32x4;
typedef __attribute__((ext_vector_type(8)))  unsigned u32x8;

union Frag16 { v16bf v; v8bf h[2]; };

// ---------------------------------------------------------------------------
// fp32 -> bf16 elementwise convert
// ---------------------------------------------------------------------------
__global__ void cvt_f32_to_bf16(const float* __restrict__ in,
                                __bf16* __restrict__ out, int n) {
  int i = blockIdx.x * blockDim.x + threadIdx.x;
  if (i < n) out[i] = (__bf16)in[i];
}

// W_x [DINNER x 160] -> bf16 [DINNER x 192], zero padded
__global__ void cvt_pad_wx(const float* __restrict__ wx,
                           __bf16* __restrict__ out) {
  int i = blockIdx.x * blockDim.x + threadIdx.x;
  if (i < DINNER * XPROJ_PAD) {
    int k = i / XPROJ_PAD, n = i - k * XPROJ_PAD;
    out[i] = (n < XPROJ_N) ? (__bf16)wx[k * XPROJ_N + n] : (__bf16)0.0f;
  }
}

// dt_in = first 128 cols of padded xproj[ROWS x 192] -> bf16 [ROWS x 128]
__global__ void extract_dtin(const float* __restrict__ xproj,
                             __bf16* __restrict__ dtin) {
  int i = blockIdx.x * blockDim.x + threadIdx.x;
  if (i < ROWS * DTRANK) {
    int r = i >> 7, c = i & 127;
    dtin[i] = (__bf16)xproj[r * XPROJ_PAD + c];
  }
}

// ---------------------------------------------------------------------------
// bf16 WMMA GEMM:  C[M,N] = A[M,K] @ B[K,N] (+ bias[N]) (+ softplus)
// All dims are exact tile multiples (M%128==0, N%64==0, K%32==0): no guards.
// Block tile 128x64, BK=32, 256 threads = 8 waves (4x2), each wave 32x32
// as 2x2 v_wmma_f32_16x16x32_bf16.
// Software pipeline with ping-pong LDS buffers:
//   - A tile t+1 DMA'd by the Tensor Data Mover (wave 0 issues descriptor,
//     s_wait_tensorcnt after compute of tile t => TDM overlaps WMMA)
//   - B tile t+1 staged transposed by all threads (global_b128 + ds_store_b16)
//   - one __syncthreads per k-step
// ---------------------------------------------------------------------------
#define BM 128
#define BN 64
#define BK 32

__global__ __launch_bounds__(256)
void gemm_bf16_wmma(const __bf16* __restrict__ A, const __bf16* __restrict__ Bm,
                    const float* __restrict__ bias, float* __restrict__ C,
                    int M, int N, int K, int act) {
  __shared__ __attribute__((aligned(16))) __bf16 sA[2][BM][BK];
  __shared__ __attribute__((aligned(16))) __bf16 sBt[2][BN][BK];

  const int tid  = threadIdx.x;
  const int lane = tid & 31;
  const int wave = tid >> 5;
  const int wr = wave & 3;            // wave row (0..3) -> 32 rows
  const int wc = wave >> 2;           // wave col (0..1) -> 32 cols
  const int bm = blockIdx.y * BM;
  const int bn = blockIdx.x * BN;

  const int lm   = lane & 15;         // row/col within 16x16 tile
  const int half = lane >> 4;         // K-half selector
  const int kb   = half * 8;

  // LDS byte addresses (low 32 bits of generic pointer == LDS offset).
  const unsigned sA_lds[2] = { (unsigned)(size_t)(void*)&sA[0][0][0],
                               (unsigned)(size_t)(void*)&sA[1][0][0] };

  // D# group 1 is loop invariant: data_size=2B, tensor_dim0=K, tensor_dim1=M,
  // tile_dim0=32, tile_dim1=128, tensor_dim0_stride=K
  const u32x8 g1 = { 1u << 16,
                     ((unsigned)K & 0xFFFFu) << 16,
                     (((unsigned)K >> 16) & 0xFFFFu) | (((unsigned)M & 0xFFFFu) << 16),
                     (((unsigned)M >> 16) & 0xFFFFu) | (32u << 16),
                     128u,
                     (unsigned)K,
                     0u, 0u };

  // B staging: fixed (k-row, n-range) per thread; pointer strength-reduced.
  const int br  = tid >> 3;           // k row within tile (0..31)
  const int bn0 = (tid & 7) * 8;      // n start (0..56)
  const __bf16* srcB = Bm + (size_t)br * N + bn + bn0;
  const size_t bstep = (size_t)BK * N;

  auto issueA = [&](int k0, unsigned lds_addr) {
    unsigned long long ga =
        (unsigned long long)(size_t)(A + (size_t)bm * K + k0);
    u32x4 g0 = { 1u,
                 lds_addr,
                 (unsigned)ga,
                 (((unsigned)(ga >> 32)) & 0x01FFFFFFu) | 0x80000000u };
    asm volatile("tensor_load_to_lds %0, %1" :: "s"(g0), "s"(g1) : "memory");
  };
  auto stageB = [&](const __bf16* src, int b) {
    v8bf bv = *(const v8bf*)src;
#pragma unroll
    for (int i = 0; i < 8; ++i) sBt[b][bn0 + i][br] = bv[i];
  };

  v8f acc[2][2] = {};

  // ---- prologue: stage tile 0 into buffer 0 -------------------------------
  if (wave == 0) issueA(0, sA_lds[0]);
  stageB(srcB, 0);
  srcB += bstep;
  if (wave == 0) __builtin_amdgcn_s_wait_tensorcnt(0);
  __syncthreads();

  int buf = 0;
  for (int k0 = 0; k0 < K; k0 += BK) {
    const int nb = buf ^ 1;
    const bool more = (k0 + BK) < K;

    // ---- stage tile t+1 into the alternate buffer (overlaps compute) -----
    if (more) {
      if (wave == 0) issueA(k0 + BK, sA_lds[nb]);
      stageB(srcB, nb);
      srcB += bstep;
    }

    // ---- compute tile t: fragments + 4 WMMA -------------------------------
#pragma unroll
    for (int i = 0; i < 2; ++i) {
      Frag16 af;
      int mrow = wr * 32 + i * 16 + lm;
      af.h[0] = *(const v8bf*)&sA[buf][mrow][kb];
      af.h[1] = *(const v8bf*)&sA[buf][mrow][kb + 16];
#pragma unroll
      for (int j = 0; j < 2; ++j) {
        Frag16 bf;
        int ncol = wc * 32 + j * 16 + lm;
        bf.h[0] = *(const v8bf*)&sBt[buf][ncol][kb];
        bf.h[1] = *(const v8bf*)&sBt[buf][ncol][kb + 16];
        acc[i][j] = __builtin_amdgcn_wmma_f32_16x16x32_bf16(
            false, af.v, false, bf.v, (short)0, acc[i][j], false, false);
      }
    }

    if (more && wave == 0) __builtin_amdgcn_s_wait_tensorcnt(0);
    __syncthreads();
    buf = nb;
  }

  // ---- epilogue: C/D layout: VGPR r -> M = r + 8*half, N = lane%16 --------
#pragma unroll
  for (int i = 0; i < 2; ++i) {
#pragma unroll
    for (int j = 0; j < 2; ++j) {
      int gcol = bn + wc * 32 + j * 16 + lm;
      float bv = bias ? bias[gcol] : 0.0f;
#pragma unroll
      for (int r = 0; r < 8; ++r) {
        int grow = bm + wr * 32 + i * 16 + r + 8 * half;
        float v = acc[i][j][r] + bv;
        if (act == 1) v = (v > 20.0f) ? v : log1pf(__expf(v));
        C[(size_t)grow * N + gcol] = v;
      }
    }
  }
}

// ---------------------------------------------------------------------------
// causal depthwise conv (K=4) + SiLU over the `hidden` half of proj.
// proj: [ROWS][2*DINNER]; writes conv fp32 + bf16 copies [ROWS][DINNER].
// ---------------------------------------------------------------------------
__global__ void conv_silu_kernel(const float* __restrict__ proj,
                                 const float* __restrict__ conv_w,
                                 const float* __restrict__ conv_b,
                                 float* __restrict__ convf,
                                 __bf16* __restrict__ convbf) {
  int idx = blockIdx.x * blockDim.x + threadIdx.x;   // over ROWS*DINNER
  if (idx >= ROWS * DINNER) return;
  int d   = idx & (DINNER - 1);
  int row = idx >> 12;               // DINNER = 4096 = 2^12
  int l   = row & (L_ - 1);
  float acc = conv_b[d];
#pragma unroll
  for (int k = 0; k < KCONV; ++k) {
    int lk = l - (KCONV - 1) + k;
    if (lk >= 0)
      acc += conv_w[k * DINNER + d] *
             proj[(size_t)(row - (KCONV - 1) + k) * (2 * DINNER) + d];
  }
  float s = acc / (1.0f + __expf(-acc));   // SiLU
  convf[idx]  = s;
  convbf[idx] = (__bf16)s;
}

// ---------------------------------------------------------------------------
// selective scan: one thread per (b,d) channel; 16-wide state in registers.
// Fuses  y = (scan + x*D) * silu(gate), writes bf16 for the output GEMM.
// ---------------------------------------------------------------------------
__global__ __launch_bounds__(256)
void scan_kernel(const float* __restrict__ delta,
                 const float* __restrict__ convf,
                 const float* __restrict__ xproj,
                 const float* __restrict__ proj,
                 const float* __restrict__ a_log,
                 const float* __restrict__ d_param,
                 __bf16* __restrict__ ybf) {
  int gid = blockIdx.x * blockDim.x + threadIdx.x;   // 0 .. B_*DINNER-1
  if (gid >= B_ * DINNER) return;
  int b = gid >> 12;                 // DINNER = 4096
  int d = gid & (DINNER - 1);

  float Arow[NSTATE], h[NSTATE];
#pragma unroll
  for (int n = 0; n < NSTATE; ++n) {
    Arow[n] = -__expf(a_log[d * NSTATE + n]);
    h[n] = 0.0f;
  }
  float Dp = d_param[d];

  for (int l = 0; l < L_; ++l) {
    int row = b * L_ + l;
    float dt = delta[(size_t)row * DINNER + d];
    float x  = convf[(size_t)row * DINNER + d];
    const float* bc = xproj + (size_t)row * XPROJ_PAD + DTRANK;  // B then C
    float dx = dt * x;
    float y = 0.0f;
#pragma unroll
    for (int n = 0; n < NSTATE; ++n) {
      float hn = __expf(dt * Arow[n]) * h[n] + dx * bc[n];
      h[n] = hn;
      y += hn * bc[NSTATE + n];
    }
    float g  = proj[(size_t)row * (2 * DINNER) + DINNER + d];
    float sg = g / (1.0f + __expf(-g));
    y = (y + x * Dp) * sg;
    ybf[(size_t)row * DINNER + d] = (__bf16)y;
  }
}

// ---------------------------------------------------------------------------
// host-side orchestration
// ---------------------------------------------------------------------------
static inline size_t align_up(size_t x, size_t a) { return (x + a - 1) & ~(a - 1); }

extern "C" void kernel_launch(void* const* d_in, const int* in_sizes, int n_in,
                              void* d_out, int out_size, void* d_ws, size_t ws_size,
                              hipStream_t stream) {
  (void)in_sizes; (void)n_in; (void)out_size; (void)ws_size;

  const float* hs      = (const float*)d_in[0];   // [B,L,DMODEL]
  const float* W_in    = (const float*)d_in[1];   // [DMODEL, 2*DINNER]
  const float* b_in    = (const float*)d_in[2];
  const float* conv_w  = (const float*)d_in[3];   // [K, DINNER]
  const float* conv_b  = (const float*)d_in[4];
  const float* W_x     = (const float*)d_in[5];   // [DINNER, 160]
  const float* W_dt    = (const float*)d_in[6];   // [DTRANK, DINNER]
  const float* b_dt    = (const float*)d_in[7];
  const float* W_out   = (const float*)d_in[8];   // [DINNER, DMODEL]
  const float* b_out   = (const float*)d_in[9];
  const float* a_log   = (const float*)d_in[10];  // [DINNER, N]
  const float* d_param = (const float*)d_in[11];  // [DINNER]
  float* out = (float*)d_out;                     // [B,L,DMODEL]

  // workspace carve-out
  char* ws = (char*)d_ws;
  size_t off = 0;
  auto carve = [&](size_t bytes) {
    off = align_up(off, 256);
    void* p = ws + off;
    off += bytes;
    return p;
  };
  __bf16* hs_bf   = (__bf16*)carve((size_t)ROWS * DMODEL * 2);
  __bf16* win_bf  = (__bf16*)carve((size_t)DMODEL * 2 * DINNER * 2);
  __bf16* wx_bf   = (__bf16*)carve((size_t)DINNER * XPROJ_PAD * 2);
  __bf16* wdt_bf  = (__bf16*)carve((size_t)DTRANK * DINNER * 2);
  __bf16* wout_bf = (__bf16*)carve((size_t)DINNER * DMODEL * 2);
  float*  proj    = (float*)carve((size_t)ROWS * 2 * DINNER * 4);
  float*  convf   = (float*)carve((size_t)ROWS * DINNER * 4);
  __bf16* convbf  = (__bf16*)carve((size_t)ROWS * DINNER * 2);
  float*  xproj   = (float*)carve((size_t)ROWS * XPROJ_PAD * 4);
  __bf16* dtin_bf = (__bf16*)carve((size_t)ROWS * DTRANK * 2);
  float*  delta   = (float*)carve((size_t)ROWS * DINNER * 4);
  __bf16* ybf     = (__bf16*)carve((size_t)ROWS * DINNER * 2);

  const int T = 256;
  auto blocks = [](size_t n, int t) { return (unsigned)((n + t - 1) / t); };

  // 1) converts (weights + activations to bf16; W_x padded to 192 cols)
  cvt_f32_to_bf16<<<blocks((size_t)ROWS * DMODEL, T), T, 0, stream>>>(hs, hs_bf, ROWS * DMODEL);
  cvt_f32_to_bf16<<<blocks((size_t)DMODEL * 2 * DINNER, T), T, 0, stream>>>(W_in, win_bf, DMODEL * 2 * DINNER);
  cvt_pad_wx<<<blocks((size_t)DINNER * XPROJ_PAD, T), T, 0, stream>>>(W_x, wx_bf);
  cvt_f32_to_bf16<<<blocks((size_t)DTRANK * DINNER, T), T, 0, stream>>>(W_dt, wdt_bf, DTRANK * DINNER);
  cvt_f32_to_bf16<<<blocks((size_t)DINNER * DMODEL, T), T, 0, stream>>>(W_out, wout_bf, DINNER * DMODEL);

  // 2) GEMM1: proj = hs @ W_in + b_in   [4096 x 8192], K=2048
  {
    dim3 grid((2 * DINNER) / BN, ROWS / BM);
    gemm_bf16_wmma<<<grid, 256, 0, stream>>>(hs_bf, win_bf, b_in, proj,
                                             ROWS, 2 * DINNER, DMODEL, 0);
  }

  // 3) conv + silu
  conv_silu_kernel<<<blocks((size_t)ROWS * DINNER, T), T, 0, stream>>>(
      proj, conv_w, conv_b, convf, convbf);

  // 4) GEMM2: xproj = conv @ W_x   [4096 x 192 padded], K=4096 (no bias)
  {
    dim3 grid(XPROJ_PAD / BN, ROWS / BM);
    gemm_bf16_wmma<<<grid, 256, 0, stream>>>(convbf, wx_bf, nullptr, xproj,
                                             ROWS, XPROJ_PAD, DINNER, 0);
  }

  // 5) dt_in -> bf16
  extract_dtin<<<blocks((size_t)ROWS * DTRANK, T), T, 0, stream>>>(xproj, dtin_bf);

  // 6) GEMM3: delta = softplus(dt_in @ W_dt + b_dt)   [4096 x 4096], K=128
  {
    dim3 grid(DINNER / BN, ROWS / BM);
    gemm_bf16_wmma<<<grid, 256, 0, stream>>>(dtin_bf, wdt_bf, b_dt, delta,
                                             ROWS, DINNER, DTRANK, 1);
  }

  // 7) selective scan (+ x*D + silu-gate), writes y as bf16
  scan_kernel<<<blocks((size_t)B_ * DINNER, T), T, 0, stream>>>(
      delta, convf, xproj, proj, a_log, d_param, ybf);

  // 8) GEMM4: out = y @ W_out + b_out   [4096 x 2048], K=4096
  {
    dim3 grid(DMODEL / BN, ROWS / BM);
    gemm_bf16_wmma<<<grid, 256, 0, stream>>>(ybf, wout_bf, b_out, out,
                                             ROWS, DMODEL, DINNER, 0);
  }
}
